// FactoredCausalSelfAttentionALiBi_19945828123060
// MI455X (gfx1250) — matrix operand order repaired and checked by reading, not compiled, not run on hardware
//
#include <hip/hip_runtime.h>
#include <hip/hip_bf16.h>
#include <math.h>

// MI455X / gfx1250, wave32. Matrix work on v_wmma_f32_16x16x32_f16.

typedef __attribute__((ext_vector_type(16))) _Float16 v16h;
typedef __attribute__((ext_vector_type(8)))  _Float16 v8h;
typedef __attribute__((ext_vector_type(8)))  float    v8f;

#define B_   2
#define T_   2048
#define C_   768
#define H_   12
#define HD_  64
#define NQK  1536           // 2*C
#define NEG_BIG (-3.0e38f)  // finite "-inf": avoids inf-inf NaN in online softmax

__device__ __forceinline__ v8f wmma_f16(v16h a, v16h b, v8f c) {
  // D = A(16x32 f16) * B(32x16 f16) + C(16x16 f32)
  return __builtin_amdgcn_wmma_f32_16x16x32_f16(
      /*neg_a=*/false, a, /*neg_b=*/false, b,
      /*c_mod=*/(short)0, c, /*reuse_a=*/false, /*reuse_b=*/false);
}

// C/D layout: VGPR r holds row M=r+8*hh, col N=lane&15. A row's 16 values live
// in one 16-lane half -> xor-shuffle reduction with masks 1,2,4,8 stays in-half.
__device__ __forceinline__ float half_max16(float v) {
  #pragma unroll
  for (int off = 1; off <= 8; off <<= 1) v = fmaxf(v, __shfl_xor(v, off, 32));
  return v;
}
__device__ __forceinline__ float half_sum16(float v) {
  #pragma unroll
  for (int off = 1; off <= 8; off <<= 1) v += __shfl_xor(v, off, 32);
  return v;
}

// ---------------------------------------------------------------------------
// Kernel 1: qk = x @ W_attn^T + b_attn, split into Q/K, stored f16 (B,H,T,hd)
// One 16x16 tile per wave. 8 waves / block. 24576 tiles total.
// A-operand (lane m=lane&15 is row; hh=lane>>4): K runs [8hh..8hh+7],[16+8hh..+7]
// B-operand (lane n=lane&15 is col): K runs [8hh..8hh+7],[16+8hh..+7]  (B[k][n]=W[n][k])
// ---------------------------------------------------------------------------
__global__ __launch_bounds__(256) void qk_proj_kernel(
    const float* __restrict__ X, const float* __restrict__ W,
    const float* __restrict__ bias,
    _Float16* __restrict__ Qh, _Float16* __restrict__ Kh) {
  const int lane = threadIdx.x & 31;
  const int wid  = threadIdx.x >> 5;
  const int wt   = blockIdx.x * 8 + wid;        // tile id
  const int tileN = wt % (NQK / 16);
  const int tileM = wt / (NQK / 16);
  const int m0 = tileM * 16, n0 = tileN * 16;
  const int m = lane & 15, hh = lane >> 4;
  const int col = n0 + m;

  const float* arow = X + (size_t)(m0 + m) * C_;
  const float* brow = W + (size_t)col * C_;

  v8f acc = {};
  for (int k0 = 0; k0 < C_; k0 += 32) {
    v16h a, bmat;
    #pragma unroll
    for (int e = 0; e < 8; ++e) {
      a[e]     = (_Float16)arow[k0 + 8 * hh + e];
      a[e + 8] = (_Float16)arow[k0 + 16 + 8 * hh + e];
    }
    #pragma unroll
    for (int e = 0; e < 8; ++e) {
      bmat[e]     = (_Float16)brow[k0 + 8 * hh + e];
      bmat[e + 8] = (_Float16)brow[k0 + 16 + 8 * hh + e];
    }
    acc = wmma_f16(a, bmat, acc);
  }

  const float bv = bias[col];
  #pragma unroll
  for (int r = 0; r < 8; ++r) {
    const int rowg = m0 + r + 8 * hh;           // flat b*T + t
    const int bb = rowg >> 11;                  // T = 2048
    const int t  = rowg & (T_ - 1);
    const float val = acc[r] + bv;
    if (col < C_) {
      const int head = col >> 6, d = col & 63;
      Qh[(((size_t)(bb * H_ + head)) * T_ + t) * HD_ + d] = (_Float16)val;
    } else {
      const int cc = col - C_;
      const int head = cc >> 6, d = cc & 63;
      Kh[(((size_t)(bb * H_ + head)) * T_ + t) * HD_ + d] = (_Float16)val;
    }
  }
}

// ---------------------------------------------------------------------------
// Kernel 2: V head-mix: v[b,h,t,d] = sum_j v_tmp[h,j]*xt[b,t,j*64+d]
// Stored TRANSPOSED f16 as Vt[b,h,d,t] so attention's P*V B-operand is a
// contiguous v16h load along the key axis.
// ---------------------------------------------------------------------------
__global__ __launch_bounds__(256) void v_mix_kernel(
    const float* __restrict__ XT, const float* __restrict__ vtmp,
    _Float16* __restrict__ Vt) {
  __shared__ float vs[H_ * H_];
  if (threadIdx.x < H_ * H_) vs[threadIdx.x] = vtmp[threadIdx.x];
  __syncthreads();
  const int idx = blockIdx.x * 256 + threadIdx.x;   // over B*T*HD = 262144
  const int d  = idx & 63;
  const int t  = (idx >> 6) & (T_ - 1);
  const int bb = idx >> 17;
  float xv[H_];
  #pragma unroll
  for (int j = 0; j < H_; ++j)
    xv[j] = XT[((size_t)(bb * T_ + t)) * C_ + j * HD_ + d];
  #pragma unroll
  for (int i = 0; i < H_; ++i) {
    float acc = 0.f;
    #pragma unroll
    for (int j = 0; j < H_; ++j) acc = fmaf(vs[i * H_ + j], xv[j], acc);
    Vt[(((size_t)(bb * H_ + i)) * HD_ + d) * T_ + t] = (_Float16)acc;
  }
}

// ---------------------------------------------------------------------------
// Kernel 3: causal ALiBi flash attention.
// One wave per (b,h,16-query tile); 8 waves/block; 3072 waves total.
// Key tiles of 32 (two 16-wide S subtiles). Online softmax state is kept
// replicated across each 16-lane half (rows r+8*hh). P transposed C->A layout
// through per-wave LDS (16x40 f16, 16B-aligned 8-element runs).
// ---------------------------------------------------------------------------
__global__ __launch_bounds__(256) void flash_attn_kernel(
    const _Float16* __restrict__ Qh, const _Float16* __restrict__ Kh,
    const _Float16* __restrict__ Vt, float* __restrict__ Y) {
  __shared__ _Float16 plds[8][16 * 40];
  const int lane = threadIdx.x & 31;
  const int wid  = threadIdx.x >> 5;
  const int wv = blockIdx.x * 8 + wid;          // 0 .. 3071
  const int qt = wv & 127;                      // T/16 = 128 query tiles
  const int h  = (wv >> 7) % H_;
  const int bb = wv / (128 * H_);
  const int m = lane & 15, hh = lane >> 4;

  const _Float16* Qp = Qh + (((size_t)(bb * H_ + h)) * T_) * HD_;
  const _Float16* Kp = Kh + (((size_t)(bb * H_ + h)) * T_) * HD_;
  const _Float16* Vp = Vt + (((size_t)(bb * H_ + h)) * HD_) * T_;

  // Q in A-operand layout, two 32-wide hd chunks
  const _Float16* qrow = Qp + (size_t)(qt * 16 + m) * HD_;
  v16h qa0 = *(const v16h*)(qrow + 16 * hh);
  v16h qa1 = *(const v16h*)(qrow + 32 + 16 * hh);

  float mrow[8], lrow[8];
  #pragma unroll
  for (int r = 0; r < 8; ++r) { mrow[r] = NEG_BIG; lrow[r] = 0.f; }
  v8f o0 = {}, o1 = {}, o2 = {}, o3 = {};

  const float scale = 0.125f;                   // 1/sqrt(64)
  // ALiBi slopes for H=12: heads 0-7: 2^-(h+1); heads 8-11: 2^(-0.5*(h-7))
  const float slope = (h < 8) ? exp2f(-(float)(h + 1)) : exp2f(-0.5f * (float)(h - 7));
  const int q_hi = qt * 16 + 15;
  _Float16* pl = &plds[wid][0];

  for (int kbase = 0; kbase <= q_hi; kbase += 32) {
    // ---- S = Q K^T for two 16-key subtiles (B[k][n] = K[key=n][k]) ----
    const int key0 = kbase + m;
    const int key1 = kbase + 16 + m;
    v16h kb00 = *(const v16h*)(Kp + (size_t)key0 * HD_ + 16 * hh);
    v16h kb01 = *(const v16h*)(Kp + (size_t)key0 * HD_ + 32 + 16 * hh);
    v16h kb10 = *(const v16h*)(Kp + (size_t)key1 * HD_ + 16 * hh);
    v16h kb11 = *(const v16h*)(Kp + (size_t)key1 * HD_ + 32 + 16 * hh);
    if (kbase + 32 <= q_hi)
      __builtin_prefetch(Kp + (size_t)(kbase + 32 + m) * HD_, 0, 3);

    v8f s0 = {}, s1 = {};
    s0 = wmma_f16(qa0, kb00, s0);
    s0 = wmma_f16(qa1, kb01, s0);
    s1 = wmma_f16(qa0, kb10, s1);
    s1 = wmma_f16(qa1, kb11, s1);

    // ---- scale + ALiBi + causal mask + online softmax ----
    float p0[8], p1[8], fac[8];
    #pragma unroll
    for (int r = 0; r < 8; ++r) {
      const int qrow_g = qt * 16 + r + 8 * hh;
      const int kc0 = kbase + m;
      const int kc1 = kbase + 16 + m;
      const float sv0 = (kc0 <= qrow_g)
          ? (s0[r] * scale + slope * (float)(kc0 - qrow_g)) : NEG_BIG;
      const float sv1 = (kc1 <= qrow_g)
          ? (s1[r] * scale + slope * (float)(kc1 - qrow_g)) : NEG_BIG;
      const float mt   = half_max16(fmaxf(sv0, sv1));
      const float mnew = fmaxf(mrow[r], mt);
      fac[r] = __expf(mrow[r] - mnew);
      p0[r]  = __expf(sv0 - mnew);
      p1[r]  = __expf(sv1 - mnew);
      lrow[r] = lrow[r] * fac[r] + half_sum16(p0[r] + p1[r]);
      mrow[r] = mnew;
    }

    // rescale O accumulators and stage P (C-layout) into LDS
    #pragma unroll
    for (int r = 0; r < 8; ++r) {
      o0[r] *= fac[r]; o1[r] *= fac[r]; o2[r] *= fac[r]; o3[r] *= fac[r];
      pl[(r + 8 * hh) * 40 + m]      = (_Float16)p0[r];
      pl[(r + 8 * hh) * 40 + 16 + m] = (_Float16)p1[r];
    }
    asm volatile("s_wait_dscnt 0" ::: "memory");   // LDS write -> read in-wave

    // re-read P in A-operand layout (row m, K run [16hh..16hh+15])
    const _Float16* prow = pl + m * 40;
    v8h pal = *(const v8h*)(prow + 16 * hh);
    v8h pah = *(const v8h*)(prow + 16 * hh + 8);
    v16h pa = __builtin_shufflevector(pal, pah, 0,1,2,3,4,5,6,7,8,9,10,11,12,13,14,15);

    // ---- O += P * V ; Vt is [d][t] so B[k][n]=V[key][dim] is contiguous ----
    const _Float16* vcol = Vp + (size_t)m * T_ + kbase + 16 * hh;
    v16h vb0 = *(const v16h*)(vcol);
    v16h vb1 = *(const v16h*)(vcol + 16 * T_);
    v16h vb2 = *(const v16h*)(vcol + 32 * T_);
    v16h vb3 = *(const v16h*)(vcol + 48 * T_);
    o0 = wmma_f16(pa, vb0, o0);
    o1 = wmma_f16(pa, vb1, o1);
    o2 = wmma_f16(pa, vb2, o2);
    o3 = wmma_f16(pa, vb3, o3);
  }

  // ---- epilogue: divide by softmax denominator, store y (B,H,T,hd) f32 ----
  #pragma unroll
  for (int r = 0; r < 8; ++r) {
    const float inv = 1.0f / lrow[r];
    const size_t rbase =
        (((size_t)(bb * H_ + h)) * T_ + (size_t)(qt * 16 + r + 8 * hh)) * HD_;
    Y[rbase +  0 + m] = o0[r] * inv;
    Y[rbase + 16 + m] = o1[r] * inv;
    Y[rbase + 32 + m] = o2[r] * inv;
    Y[rbase + 48 + m] = o3[r] * inv;
  }
}

// ---------------------------------------------------------------------------
// Kernel 4: output head-mix: out[b,t,i*64+d] = sum_j proj_tmp[i,j]*y[b,j,t,d]
// ---------------------------------------------------------------------------
__global__ __launch_bounds__(256) void out_proj_kernel(
    const float* __restrict__ Y, const float* __restrict__ ptmp,
    float* __restrict__ out) {
  __shared__ float ps[H_ * H_];
  if (threadIdx.x < H_ * H_) ps[threadIdx.x] = ptmp[threadIdx.x];
  __syncthreads();
  const int idx = blockIdx.x * 256 + threadIdx.x;   // over B*T*HD
  const int d  = idx & 63;
  const int t  = (idx >> 6) & (T_ - 1);
  const int bb = idx >> 17;
  float yv[H_];
  #pragma unroll
  for (int j = 0; j < H_; ++j)
    yv[j] = Y[(((size_t)(bb * H_ + j)) * T_ + t) * HD_ + d];
  #pragma unroll
  for (int i = 0; i < H_; ++i) {
    float acc = 0.f;
    #pragma unroll
    for (int j = 0; j < H_; ++j) acc = fmaf(ps[i * H_ + j], yv[j], acc);
    out[((size_t)(bb * T_ + t)) * C_ + i * HD_ + d] = acc;
  }
}

// ---------------------------------------------------------------------------
extern "C" void kernel_launch(void* const* d_in, const int* in_sizes, int n_in,
                              void* d_out, int out_size, void* d_ws, size_t ws_size,
                              hipStream_t stream) {
  (void)in_sizes; (void)n_in; (void)out_size; (void)ws_size;
  const float* x  = (const float*)d_in[0];  // x_norm_for_qk (B,T,C)
  const float* xt = (const float*)d_in[1];  // xt_current_for_v (B,T,C)
  const float* W  = (const float*)d_in[2];  // W_attn (2C,C)
  const float* ba = (const float*)d_in[3];  // b_attn (2C)
  const float* vt = (const float*)d_in[4];  // v_tmp (H,H)
  const float* pt = (const float*)d_in[5];  // proj_tmp (H,H)
  float* out = (float*)d_out;               // (B,T,C) f32

  // Workspace: Qh, Kh, Vt (f16, 6.29MB each) + Y (f32, 12.58MB) = ~31.5MB
  const size_t nqkv = (size_t)B_ * H_ * T_ * HD_;  // 3,145,728 elements
  _Float16* Qh = (_Float16*)d_ws;
  _Float16* Kh = Qh + nqkv;
  _Float16* Vt = Kh + nqkv;
  float*    Y  = (float*)(Vt + nqkv);

  // 24576 output tiles / 8 waves per block
  qk_proj_kernel<<<3072, 256, 0, stream>>>(x, W, ba, Qh, Kh);
  // B*T*HD = 262144 threads
  v_mix_kernel<<<1024, 256, 0, stream>>>(xt, vt, Vt);
  // B*H*(T/16) = 3072 waves / 8 per block
  flash_attn_kernel<<<384, 256, 0, stream>>>(Qh, Kh, Vt, Y);
  out_proj_kernel<<<1024, 256, 0, stream>>>(Y, pt, out);
}
